// MyModel_87522843558558
// MI455X (gfx1250) — matrix operand order repaired and checked by reading, not compiled, not run on hardware
//
#include <hip/hip_runtime.h>

// ---------------------------------------------------------------------------
// Linear RNN scan:  state_t = state_{t-1} @ W_s + x_t @ W_x + b
//   Kernel 1 (parallel): out[b,t,:] = x[b,t,:] @ W_x + b        (bf16 WMMA)
//   Kernel 2 (persistent single WG): out[b,t,:] += state @ W_s  (bf16 WMMA,
//            state kept in LDS as bf16, W_s B-fragments in registers)
// ---------------------------------------------------------------------------

typedef __attribute__((ext_vector_type(16))) __bf16 v16bf;
typedef __attribute__((ext_vector_type(8)))  __bf16 v8bf;
typedef __attribute__((ext_vector_type(16))) float  v16f;
typedef __attribute__((ext_vector_type(8)))  float  v8f;

#define RNN_B 32
#define RNN_T 2048
#define RNN_F 256
#define RNN_S 256

static __device__ __forceinline__ v8bf cvt8(v8f f) {
  return __builtin_convertvector(f, v8bf);   // -> packed v_cvt_pk_bf16_f32
}

static __device__ __forceinline__ v16bf join8(v8bf a, v8bf b) {
  return __builtin_shufflevector(a, b, 0, 1, 2, 3, 4, 5, 6, 7,
                                       8, 9, 10, 11, 12, 13, 14, 15);
}

// ---------------------------------------------------------------------------
// Kernel 1: U = X @ W_x + b, written into out[B*T, S] (== out[B,T,S]).
// Grid: (512, 16) x 256 threads. Each wave computes one 16x16 tile, K=256.
// ---------------------------------------------------------------------------
__global__ __launch_bounds__(256)
void rnn_ugemm_kernel(const float* __restrict__ x, const float* __restrict__ W,
                      const float* __restrict__ bias, float* __restrict__ out) {
  __shared__ __bf16 Wt[16][RNN_F + 8];  // W_x^T tile, bf16, padded rows

  const int lane = threadIdx.x & 31;
  const int wave = threadIdx.x >> 5;
  const int n0   = blockIdx.y << 4;
  const int m0   = (blockIdx.x * 8 + wave) << 4;

  // Stage W_x columns n0..n0+15 transposed into LDS as bf16 (coalesced reads).
  for (int i = threadIdx.x; i < RNN_F * 16; i += 256) {
    const int k = i >> 4, nn = i & 15;
    Wt[nn][k] = (__bf16)W[(size_t)(RNN_S + k) * RNN_S + n0 + nn];
  }
  __syncthreads();

  // B fragments: lane holds column n = lane%16, 16 consecutive K values
  // (lanes 0-15: K=kc*32+0..15, lanes 16-31: K=kc*32+16..31).
  v16bf bw[8];
  {
    const __bf16* base = &Wt[lane & 15][(lane >> 4) << 4];
    for (int kc = 0; kc < 8; ++kc) {
      const __bf16* p = base + kc * 32;
      bw[kc] = join8(*(const v8bf*)p, *(const v8bf*)(p + 8));
    }
  }

  // A fragments straight from global x (f32 -> packed bf16 cvt).
  // Lane holds row m0+lane%16; lanes 0-15 K=0..15, lanes 16-31 K=16..31.
  v8f c = {};
  const float* arow = x + (size_t)(m0 + (lane & 15)) * RNN_F + ((lane >> 4) << 4);
  for (int kc = 0; kc < 8; ++kc) {
    const float* ap = arow + kc * 32;
    v8f f0 = *(const v8f*)(ap);        // K   .. K+7
    v8f f1 = *(const v8f*)(ap + 8);    // K+8 .. K+15
    v16bf af = join8(cvt8(f0), cvt8(f1));
    c = __builtin_amdgcn_wmma_f32_16x16x32_bf16(false, af, false, bw[kc],
                                                (short)0, c, false, false);
  }

  // D layout: VGPR r -> row M = r + 8*(lane/16), col N = lane%16.
  const int   nn    = n0 + (lane & 15);
  const float bn    = bias[nn];
  const int   mbase = m0 + ((lane >> 4) << 3);
  for (int r = 0; r < 8; ++r)
    out[(size_t)(mbase + r) * RNN_S + nn] = c[r] + bn;
}

// ---------------------------------------------------------------------------
// Kernel 2: persistent scan. One workgroup, 32 waves = the 32 16x16 tiles of
// the [32 x 256] state. In-place: out[b,t,:] += state_{t-1} @ W_s, and that
// result becomes state_t (bf16 copy kept in LDS). Final state appended.
// ---------------------------------------------------------------------------
__global__ __launch_bounds__(1024)
void rnn_scan_kernel(const float* __restrict__ W, float* __restrict__ out) {
  __shared__ __bf16 st[RNN_B][RNN_S + 8];  // bf16 state, padded rows

  const int lane = threadIdx.x & 31;
  const int wave = threadIdx.x >> 5;
  const int mt   = wave >> 4;                 // 0..1   (batch tile)
  const int nt   = wave & 15;                 // 0..15  (state-col tile)
  const int n    = (nt << 4) + (lane & 15);
  const int m0   = mt << 4;

  // Preload W_s B-fragments into registers (one-time strided gather).
  v16bf bw[8];
  for (int kc = 0; kc < 8; ++kc) {
    const int kb = kc * 32 + ((lane >> 4) << 4);
    v16f wf;
    for (int i = 0; i < 16; ++i)
      wf[i] = W[(size_t)(kb + i) * RNN_S + n];
    bw[kc] = __builtin_convertvector(wf, v16bf);
  }

  // state_0 = 0
  for (int i = threadIdx.x; i < RNN_B * (RNN_S + 8); i += 1024)
    (&st[0][0])[i] = (__bf16)0.0f;
  __syncthreads();

  const __bf16* abase = &st[m0 + (lane & 15)][(lane >> 4) << 4];
  const int mb = m0 + ((lane >> 4) << 3);

  for (int t = 0; t < RNN_T; ++t) {
    // Build A fragments from LDS state (conflict-free b128 reads) and MMA.
    v8f c = {};
    for (int kc = 0; kc < 8; ++kc) {
      const __bf16* p = abase + kc * 32;
      v16bf af = join8(*(const v8bf*)p,          // K    .. K+7
                       *(const v8bf*)(p + 8));   // K+8  .. K+15
      c = __builtin_amdgcn_wmma_f32_16x16x32_bf16(false, af, false, bw[kc],
                                                  (short)0, c, false, false);
    }

    // In-place: d = state@W_s + U[t] (U pre-written in out); prefetch t+1.
    float d[8];
    for (int r = 0; r < 8; ++r) {
      const size_t idx = (size_t)(mb + r) * (RNN_T * RNN_S) + (size_t)t * RNN_S + n;
      d[r] = c[r] + out[idx];
      out[idx] = d[r];
      if (t + 1 < RNN_T) __builtin_prefetch(&out[idx + RNN_S], 0, 0);
    }

    __syncthreads();  // all A-fragment reads of state_{t-1} complete
    for (int r = 0; r < 8; ++r)
      st[mb + r][n] = (__bf16)d[r];
    if (t == RNN_T - 1)
      for (int r = 0; r < 8; ++r)
        out[(size_t)RNN_B * RNN_T * RNN_S + (size_t)(mb + r) * RNN_S + n] = d[r];
    __syncthreads();  // state_t visible to all waves
  }
}

// ---------------------------------------------------------------------------
extern "C" void kernel_launch(void* const* d_in, const int* in_sizes, int n_in,
                              void* d_out, int out_size, void* d_ws, size_t ws_size,
                              hipStream_t stream) {
  (void)in_sizes; (void)n_in; (void)out_size; (void)d_ws; (void)ws_size;
  const float* x    = (const float*)d_in[0];  // [B,T,F]
  const float* W    = (const float*)d_in[1];  // [S+F,S]
  const float* bias = (const float*)d_in[2];  // [S]
  float* out = (float*)d_out;                 // [B,T,S] ++ [B,S]

  dim3 gA((RNN_B * RNN_T / 16) / 8, RNN_S / 16, 1);  // (512, 16)
  rnn_ugemm_kernel<<<gA, 256, 0, stream>>>(x, W, bias, out);
  rnn_scan_kernel<<<1, 1024, 0, stream>>>(W, out);
}
